// SingleMofifModel_85392539779240
// MI455X (gfx1250) — compile-verified, matrix operand order checked by reading, not run on hardware
//
#include <hip/hip_runtime.h>
#include <math.h>

// Problem constants (from reference)
#define LSEQ 100
#define WW   10
#define RR   3
#define TOPK 5
#define LOUT 91          // L - W + 1
#define NTILE 6          // ceil(91 / 16)
#define WAVES 8          // waves per block (wave32 -> 256 threads)
#define NEG_INF (-3.0e38f)

typedef __attribute__((ext_vector_type(16))) _Float16 v16h;
typedef __attribute__((ext_vector_type(8)))  float    v8f;

// Raw v_rcp_f32 (avoid the IEEE div_scale/div_fmas/div_fixup expansion)
__device__ __forceinline__ float rcpf(float x) { return __builtin_amdgcn_rcpf(x); }

// mish(x) = x * tanh(softplus(x)) = x * ((1+e^x)^2 - 1) / ((1+e^x)^2 + 1)
// Clamp the exp argument so large x doesn't produce inf/inf = NaN (mish(x)->x there).
__device__ __forceinline__ float mishf(float x) {
  float e  = __expf(fminf(x, 15.0f));
  float u  = 1.0f + e;
  float u2 = u * u;
  return x * (u2 - 1.0f) * rcpf(u2 + 1.0f);
}

__global__ __launch_bounds__(256) void motif_kernel(
    const int*   __restrict__ X,   const float* __restrict__ pwm,
    const float* __restrict__ w1,  const float* __restrict__ b1,
    const float* __restrict__ W2,  const float* __restrict__ b2,
    const float* __restrict__ w3,  const float* __restrict__ b3,
    const float* __restrict__ g1,  const float* __restrict__ gb1,
    const float* __restrict__ g2,  const float* __restrict__ gb2,
    float* __restrict__ out, int B)
{
  // Block-shared constant tables
  __shared__ float s_logp[RR * WW * 4];          // log(pwm)
  __shared__ float s_w1[32], s_b1[32], s_b2[32], s_w3[32], s_gb1[32], s_g2[32];
  __shared__ float s_W2[32 * 32];
  __shared__ float s_g1[32 * TOPK];
  // Per-wave scratch
  __shared__ int   s_X[WAVES][104];              // one sequence row (100 ints, padded)
  __shared__ float s_Zs[WAVES][NTILE * 16];      // sigmoid scores per position
  __shared__ float s_y[WAVES][NTILE * 16];       // raw PWM scores per position

  const int tid  = threadIdx.x;
  const int wave = tid >> 5;
  const int lane = tid & 31;
  const int b    = blockIdx.x * WAVES + wave;

  if (b < B) __builtin_prefetch(&X[b * LSEQ], 0, 0);  // global_prefetch_b8

  // ---- stage constants (block-wide) ----
  for (int i = tid; i < RR * WW * 4; i += 256) s_logp[i] = __logf(pwm[i]);
  for (int i = tid; i < 32; i += 256) {
    s_w1[i] = w1[i]; s_b1[i] = b1[i]; s_b2[i] = b2[i];
    s_w3[i] = w3[i]; s_gb1[i] = gb1[i]; s_g2[i] = g2[i];
  }
  for (int i = tid; i < 32 * 32; i += 256)   s_W2[i] = W2[i];
  for (int i = tid; i < 32 * TOPK; i += 256) s_g1[i] = g1[i];
  // ---- stage this wave's sequence row ----
  if (b < B) {
    for (int i = lane; i < LSEQ; i += 32) s_X[wave][i] = X[b * LSEQ + i];
  }
  __syncthreads();

  const float b3v  = b3[0];
  const float gb2v = gb2[0];
  const int   hi   = lane >> 4;   // which 16-lane half
  const int   m    = lane & 15;   // A-matrix row / position-in-tile; also N index for B/C

  // W2 packed into the 16-bit B-matrix layout (32x16 per tile, K striped like A):
  // lane half hi: v16h halves 0..7 -> K = hi*8 + j, halves 8..15 -> K = 16 + hi*8 + j; N = lane%16.
  // B[k][n] = W2[o=n][h=k]  (einsum 'brlh,oh->brlo')
  // Also preload this lane's w1/b1 slices (A layout) and w3/b2 scalars (C layout)
  // into registers so the tile loop issues no redundant LDS traffic.
  v16h  bmat0, bmat1;
  float w1r[16], b1r[16];
  #pragma unroll
  for (int j = 0; j < 8; ++j) {
    int k0 = hi * 8 + j;
    int k1 = 16 + hi * 8 + j;
    bmat0[j]     = (_Float16)s_W2[(m)      * 32 + k0];
    bmat0[j + 8] = (_Float16)s_W2[(m)      * 32 + k1];
    bmat1[j]     = (_Float16)s_W2[(m + 16) * 32 + k0];
    bmat1[j + 8] = (_Float16)s_W2[(m + 16) * 32 + k1];
    w1r[j]     = s_w1[k0];  b1r[j]     = s_b1[k0];
    w1r[j + 8] = s_w1[k1];  b1r[j + 8] = s_b1[k1];
  }
  const float w3n0 = s_w3[m],  w3n1 = s_w3[m + 16];
  const float b2n0 = s_b2[m],  b2n1 = s_b2[m + 16];

  float sr[RR], llr[RR];

  #pragma unroll
  for (int r = 0; r < RR; ++r) {
    const float* lp = &s_logp[r * WW * 4];

    for (int t = 0; t < NTILE; ++t) {
      // Both halves of the wave handle the same 16 positions (required by A layout).
      int l  = t * 16 + m;
      int lc = (l < LOUT) ? l : (LOUT - 1);     // clamp tail-tile reads
      float y = 0.0f;
      #pragma unroll
      for (int w = 0; w < WW; ++w) y += lp[w * 4 + s_X[wave][lc + w]];
      y *= (1.0f / WW);

      // h1 = mish(y*w1 + b1), computed directly in WMMA A-register layout.
      v16h a;
      #pragma unroll
      for (int j = 0; j < 16; ++j)
        a[j] = (_Float16)mishf(y * w1r[j] + b1r[j]);

      // h2(16 pos x 32 out) = A(16x32) * W2^T via two 16x16x32 WMMAs
      v8f c0 = {}; v8f c1 = {};
      c0 = __builtin_amdgcn_wmma_f32_16x16x32_f16(false, a, false, bmat0, (short)0, c0, false, false);
      c1 = __builtin_amdgcn_wmma_f32_16x16x32_f16(false, a, false, bmat1, (short)0, c1, false, false);

      // Layer 3: z[m] = sum_n mish(h2[m][n] + b2[n]) * w3[n]
      // C layout: lanes 0-15 hold N=lane, M=j; lanes 16-31 hold N=lane-16, M=j+8.
      float z[8];
      #pragma unroll
      for (int j = 0; j < 8; ++j) {
        float p = mishf(c0[j] + b2n0) * w3n0 + mishf(c1[j] + b2n1) * w3n1;
        p += __shfl_xor(p, 1, 32);   // butterflies with mask<16 stay inside each
        p += __shfl_xor(p, 2, 32);   // 16-lane half -> per-half row sums
        p += __shfl_xor(p, 4, 32);
        p += __shfl_xor(p, 8, 32);
        z[j] = p;
      }
      // Store Zs (one lane per half) and raw y (low half lanes)
      if (m == 0) {
        #pragma unroll
        for (int j = 0; j < 8; ++j) {
          int l2 = t * 16 + hi * 8 + j;
          if (l2 < LOUT)
            s_Zs[wave][l2] = rcpf(1.0f + __expf(-(z[j] + b3v)));
        }
      }
      if (hi == 0 && l < LOUT) s_y[wave][l] = y;
    }
    __syncthreads();   // uniform across block; makes LDS scratch lane-visible

    // ---- top-5 of the 91 scores (descending, ties -> lower index, like argsort) ----
    float own[3]; int oidx[3];
    #pragma unroll
    for (int q = 0; q < 3; ++q) {
      int i = lane + 32 * q;
      oidx[q] = i;
      own[q]  = (i < LOUT) ? s_Zs[wave][i] : NEG_INF;
    }
    float zsc[TOPK]; int bestl = 0;
    #pragma unroll
    for (int ssel = 0; ssel < TOPK; ++ssel) {
      float v = own[0]; int idx = oidx[0];
      #pragma unroll
      for (int q = 1; q < 3; ++q)
        if (own[q] > v || (own[q] == v && oidx[q] < idx)) { v = own[q]; idx = oidx[q]; }
      #pragma unroll
      for (int mask = 1; mask < 32; mask <<= 1) {
        float ov = __shfl_xor(v, mask, 32);
        int   oi = __shfl_xor(idx, mask, 32);
        if (ov > v || (ov == v && oi < idx)) { v = ov; idx = oi; }
      }
      zsc[ssel] = v;
      if (ssel == 0) bestl = idx;
      #pragma unroll
      for (int q = 0; q < 3; ++q) if (oidx[q] == idx) own[q] = NEG_INF;
    }

    // ---- gating net: g = mish(g1 @ zsc + gb1); s = g . g2 + gb2 ----
    float acc = s_gb1[lane];
    #pragma unroll
    for (int k = 0; k < TOPK; ++k) acc += zsc[k] * s_g1[lane * TOPK + k];
    float gv = mishf(acc) * s_g2[lane];
    #pragma unroll
    for (int mask = 1; mask < 32; mask <<= 1) gv += __shfl_xor(gv, mask, 32);
    sr[r]  = gv + gb2v;
    llr[r] = s_y[wave][bestl];
    __syncthreads();   // protect s_Zs/s_y before the next r overwrites them
  }

  // ---- softmax over r and output ----
  if (b < B && lane == 0) {
    float mx  = fmaxf(sr[0], fmaxf(sr[1], sr[2]));
    float e0  = __expf(sr[0] - mx), e1 = __expf(sr[1] - mx), e2 = __expf(sr[2] - mx);
    float inv = rcpf(e0 + e1 + e2);
    out[b * 3 + 0] = e0 * inv;
    out[b * 3 + 1] = e1 * inv;
    out[b * 3 + 2] = e2 * inv;
    out[3 * B + b * 3 + 0] = llr[0];
    out[3 * B + b * 3 + 1] = llr[1];
    out[3 * B + b * 3 + 2] = llr[2];
  }
}

extern "C" void kernel_launch(void* const* d_in, const int* in_sizes, int n_in,
                              void* d_out, int out_size, void* d_ws, size_t ws_size,
                              hipStream_t stream) {
  const int*   X   = (const int*)  d_in[0];
  const float* pwm = (const float*)d_in[1];
  const float* w1  = (const float*)d_in[2];
  const float* b1  = (const float*)d_in[3];
  const float* W2  = (const float*)d_in[4];
  const float* b2  = (const float*)d_in[5];
  const float* w3  = (const float*)d_in[6];
  const float* b3  = (const float*)d_in[7];
  const float* g1  = (const float*)d_in[8];
  const float* gb1 = (const float*)d_in[9];
  const float* g2  = (const float*)d_in[10];
  const float* gb2 = (const float*)d_in[11];
  float* out = (float*)d_out;

  int B = in_sizes[0] / LSEQ;               // 16384
  int blocks = (B + WAVES - 1) / WAVES;     // one wave (wave32) per batch element
  motif_kernel<<<blocks, 256, 0, stream>>>(X, pwm, w1, b1, W2, b2, w3, b3,
                                           g1, gb1, g2, gb2, out, B);
}
